// VanillaRNN_35467839931013
// MI455X (gfx1250) — matrix operand order repaired and checked by reading, not compiled, not run on hardware
//
#include <hip/hip_runtime.h>

// ---------------------------------------------------------------------------
// VanillaRNN (2-layer) for gfx1250: bf16 WMMA GEMMs with async-to-LDS double
// buffering + persistent scan kernel with software grid barrier per timestep.
// ---------------------------------------------------------------------------

typedef __bf16  v16bf __attribute__((ext_vector_type(16)));
typedef float   v8f   __attribute__((ext_vector_type(8)));
typedef int     v4i   __attribute__((ext_vector_type(4)));
typedef int     v8i   __attribute__((ext_vector_type(8)));

#define AS1 __attribute__((address_space(1)))
#define AS3 __attribute__((address_space(3)))

#if __has_builtin(__builtin_amdgcn_global_load_async_to_lds_b128) && \
    __has_builtin(__builtin_amdgcn_s_wait_asynccnt)
#define HAVE_ASYNC_LDS 1
#else
#define HAVE_ASYNC_LDS 0
#endif

constexpr int B = 64, T = 512, I = 512, H = 1024;
constexpr size_t M0 = (size_t)B * T;          // 32768 rows (b*T + t)
constexpr size_t OUT1_ELEMS = M0 * H;         // 33,554,432

// ---- workspace layout (bytes; every section is 256B aligned) --------------
constexpr size_t CTRL_OFF  = 0;                                   // 2 uints
constexpr size_t XBF_OFF   = 256;                                 // x in bf16
constexpr size_t WXH0_OFF  = XBF_OFF  + M0 * I * 2;
constexpr size_t WHH0_OFF  = WXH0_OFF + (size_t)H * I * 2;
constexpr size_t WXH1_OFF  = WHH0_OFF + (size_t)H * H * 2;
constexpr size_t WHH1_OFF  = WXH1_OFF + (size_t)H * H * 2;
constexpr size_t HB0_OFF   = WHH1_OFF + (size_t)H * H * 2;        // ping-pong h, layer0
constexpr size_t HB1_OFF   = HB0_OFF  + 2ull * B * H * 2;         // ping-pong h, layer1
constexpr size_t XPROJ_OFF = HB1_OFF  + 2ull * B * H * 2;         // f32, reused L0/L1
constexpr size_t OUT0_OFF  = XPROJ_OFF + M0 * H * 4;              // layer0 outputs bf16

// ---------------------------------------------------------------------------
__device__ __forceinline__ unsigned short f2bf(float f) {
  unsigned u = __float_as_uint(f);
  unsigned r = u + 0x7FFFu + ((u >> 16) & 1u);   // round-to-nearest-even
  return (unsigned short)(r >> 16);
}

__device__ __forceinline__ v16bf make_frag(v4i q0, v4i q1) {
  v8i r = __builtin_shufflevector(q0, q1, 0, 1, 2, 3, 4, 5, 6, 7);
  return __builtin_bit_cast(v16bf, r);
}

__device__ __forceinline__ v8f wmma_bf16(v16bf a, v16bf b, v8f c) {
  return __builtin_amdgcn_wmma_f32_16x16x32_bf16(
      /*neg_a=*/false, a, /*neg_b=*/false, b,
      /*c_mod=*/(short)0, c, /*reuse_a=*/false, /*reuse_b=*/false);
}

__device__ __forceinline__ void lds_fill_b128(const unsigned short* gsrc, int* ldst) {
#if HAVE_ASYNC_LDS
  __builtin_amdgcn_global_load_async_to_lds_b128((AS1 v4i*)gsrc, (AS3 v4i*)ldst, 0, 0);
#else
  *(v4i*)ldst = *(const v4i*)gsrc;
#endif
}

// ---------------------------------------------------------------------------
__global__ void zero_ctrl_kernel(unsigned* ctrl) { ctrl[0] = 0u; ctrl[1] = 0u; }

__global__ void f32_to_bf16_kernel(const float* __restrict__ src,
                                   unsigned short* __restrict__ dst, size_t n) {
  size_t i = (size_t)blockIdx.x * blockDim.x + threadIdx.x;
  size_t stride = (size_t)gridDim.x * blockDim.x;
  for (; i < n; i += stride) dst[i] = f2bf(src[i]);
}

// ---------------------------------------------------------------------------
// C[M x H] = A[M x K](bf16) @ W[H x K]^T(bf16) + bias   (f32 out)
// Block: 256 thr (8 waves), tile 128M x 256N, K-chunks of 32 double-buffered
// in LDS via GLOBAL_LOAD_ASYNC_TO_LDS_B128 (fallback: sync copy).
// Wave (wm 0..1, wn 0..3) computes a 64x64 region = 4x4 WMMA subtiles
// (16 WMMAs per 16 ds_load_b128 per chunk). LDS use: 60KB.
// Staging addresses are fully hoisted: each thread owns 6 fixed b128 slots.
// ---------------------------------------------------------------------------
__global__ __launch_bounds__(256) void gemm_bias_kernel(
    const unsigned short* __restrict__ A, const unsigned short* __restrict__ W,
    const float* __restrict__ bias, float* __restrict__ C, int K) {
  __shared__ int lA[2][128 * 20];   // 128 rows x 32 bf16 (16 ints), stride 20 ints
  __shared__ int lB[2][256 * 20];   // 256 rows x 32 bf16

  const int tid  = threadIdx.x;
  const int wave = tid >> 5;
  const int lane = tid & 31;
  const int half = lane >> 4;
  const int l15  = lane & 15;
  const int wm = wave & 1, wn = wave >> 1;
  const int n0 = blockIdx.x * 256;
  const int m0 = blockIdx.y * 128;

  // ---- hoisted staging addresses: thread owns quad q of rows r0, r0+64,
  //      (and +128, +192 for the 256-row W tile) ----
  const int r0 = tid >> 2, q = tid & 3;
  const unsigned short* aS0 = A + (size_t)(m0 + r0) * K + q * 8;
  const unsigned short* aS1 = aS0 + (size_t)64 * K;
  const unsigned short* wS0 = W + (size_t)(n0 + r0) * K + q * 8;
  const unsigned short* wS1 = wS0 + (size_t)64 * K;
  const unsigned short* wS2 = wS0 + (size_t)128 * K;
  const unsigned short* wS3 = wS0 + (size_t)192 * K;
  const int dA0 = r0 * 20 + q * 4, dA1 = dA0 + 64 * 20;
  const int dB0 = dA0, dB1 = dB0 + 64 * 20, dB2 = dB0 + 128 * 20, dB3 = dB0 + 192 * 20;

  auto stage = [&](int k0, int buf) {
    lds_fill_b128(aS0 + k0, &lA[buf][dA0]);
    lds_fill_b128(aS1 + k0, &lA[buf][dA1]);
    lds_fill_b128(wS0 + k0, &lB[buf][dB0]);
    lds_fill_b128(wS1 + k0, &lB[buf][dB1]);
    lds_fill_b128(wS2 + k0, &lB[buf][dB2]);
    lds_fill_b128(wS3 + k0, &lB[buf][dB3]);
  };

  v8f acc[4][4] = {};
  const int nchunks = K >> 5;
  stage(0, 0);

  for (int c = 0; c < nchunks; ++c) {
    const int cur = c & 1;
#if HAVE_ASYNC_LDS
    __builtin_amdgcn_s_wait_asynccnt(0);   // my async writes into buf[cur] done
#endif
    __syncthreads();                       // everyone's staging of buf[cur] done;
                                           // everyone done reading buf[1-cur]
    if (c + 1 < nchunks) stage((c + 1) << 5, 1 - cur);

    v16bf af[4], bf[4];
#pragma unroll
    for (int i = 0; i < 4; ++i) {
      int row = wm * 64 + i * 16 + l15;
      const v4i* p = (const v4i*)&lA[cur][row * 20 + half * 4];
      af[i] = make_frag(p[0], p[2]);       // K pairs {0..7}/{16..23} (+8 if hi half)
    }
#pragma unroll
    for (int j = 0; j < 4; ++j) {
      int row = wn * 64 + j * 16 + l15;
      const v4i* p = (const v4i*)&lB[cur][row * 20 + half * 4];
      bf[j] = make_frag(p[0], p[2]);       // K pairs {0..7}/{16..23} (+8 if hi half)
    }
#pragma unroll
    for (int i = 0; i < 4; ++i)
#pragma unroll
      for (int j = 0; j < 4; ++j)
        acc[i][j] = wmma_bf16(af[i], bf[j], acc[i][j]);
  }

#pragma unroll
  for (int j = 0; j < 4; ++j) {
    int ng = n0 + wn * 64 + j * 16 + l15;
    float bb = bias[ng];
#pragma unroll
    for (int i = 0; i < 4; ++i) {
#pragma unroll
      for (int r = 0; r < 8; ++r) {
        int mg = m0 + wm * 64 + i * 16 + r + half * 8;
        C[(size_t)mg * H + ng] = acc[i][j][r] + bb;
      }
    }
  }
}

// ---------------------------------------------------------------------------
// Software grid barrier for the persistent scan kernel (32 blocks).
// ---------------------------------------------------------------------------
__device__ __forceinline__ void grid_sync(unsigned* cnt, unsigned* gen, unsigned nb) {
  __threadfence();                // make this wave's h stores device-visible
  __syncthreads();
  if (threadIdx.x == 0) {
    unsigned old = __hip_atomic_load(gen, __ATOMIC_RELAXED, __HIP_MEMORY_SCOPE_AGENT);
    unsigned prev = __hip_atomic_fetch_add(cnt, 1u, __ATOMIC_ACQ_REL, __HIP_MEMORY_SCOPE_AGENT);
    if (prev == nb - 1u) {
      __hip_atomic_store(cnt, 0u, __ATOMIC_RELEASE, __HIP_MEMORY_SCOPE_AGENT);
      __hip_atomic_fetch_add(gen, 1u, __ATOMIC_RELEASE, __HIP_MEMORY_SCOPE_AGENT);
    } else {
      while (__hip_atomic_load(gen, __ATOMIC_RELAXED, __HIP_MEMORY_SCOPE_AGENT) == old)
        __builtin_amdgcn_s_sleep(2);
    }
  }
  __syncthreads();
  __threadfence();                // invalidate near caches before reading new h
}

// ---------------------------------------------------------------------------
// Persistent recurrent scan: h_t = tanh(xproj[:,t,:] + h_{t-1} @ Whh^T).
// 32 blocks x 8 waves = 256 waves; wave gw owns output tile (mt = gw&3,
// nt = gw>>2) of the 64x1024 state. K=1024 -> 32 bf16 WMMAs per wave per step.
// Whh (2MB bf16) stays L2-resident across all 512 steps; xproj streams from
// HBM/L2 with NT loads + prefetch of the next timestep.
// ---------------------------------------------------------------------------
template <bool OUT_BF16>
__global__ __launch_bounds__(256) void rnn_scan_kernel(
    const float* __restrict__ xproj,           // [M0, H] f32, rows = b*T + t
    const unsigned short* __restrict__ Whh,    // [H, H] bf16
    unsigned short* __restrict__ hbuf,         // [2][B*H] bf16 ping-pong, [0] = h_init
    void* __restrict__ out_seq,                // bf16 (layer0) or f32 (layer1), [M0, H]
    float* __restrict__ hn_out,                // [B, H] f32 (final state)
    unsigned* __restrict__ ctrl, int nblocks) {
  const int wave = threadIdx.x >> 5;
  const int lane = threadIdx.x & 31;
  const int half = lane >> 4;
  const int l15  = lane & 15;
  const int gw = blockIdx.x * 8 + wave;        // 0..255
  const int mt = gw & 3;                       // batch tile (4 x 16 rows)
  const int nt = gw >> 2;                      // hidden tile (64 x 16 cols)
  const int n  = nt * 16 + l15;

  const unsigned short* bptr = Whh + (size_t)(nt * 16 + l15) * H + half * 8;
  const int aoff  = (mt * 16 + l15) * H + half * 8;     // h_prev fragment offset
  const int b0r   = mt * 16 + half * 8;                 // batch row of acc elem r=0
  const size_t hbase = (size_t)b0r * H + n;             // h state offset (+ r*H)
  size_t xoff = (size_t)b0r * T * H + n;                // xproj/out offset (+ r*T*H), += H per t

  for (int t = 0; t < T; ++t) {
    const unsigned short* aptr = hbuf + (size_t)(t & 1) * (B * H) + aoff;
    unsigned short* hnext      = hbuf + (size_t)((t + 1) & 1) * (B * H);

    // Hide next timestep's xproj (HBM-resident stream) behind this step's WMMAs.
    if (t + 1 < T) {
#pragma unroll
      for (int r = 0; r < 8; ++r)
        __builtin_prefetch(xproj + xoff + (size_t)r * (T * H) + H, 0, 1);
    }

    v8f acc = {};
#pragma unroll 8
    for (int k0 = 0; k0 < H; k0 += 32) {
      v4i a0 = *(const v4i*)(aptr + k0);        // K {0..7}  (+8 if hi half)
      v4i a1 = *(const v4i*)(aptr + k0 + 16);   // K {16..23}(+8 if hi half)
      v4i b0 = *(const v4i*)(bptr + k0);        // K {0..7}  (+8 if hi half)
      v4i b1 = *(const v4i*)(bptr + k0 + 16);   // K {16..23}(+8 if hi half)
      acc = wmma_bf16(make_frag(a0, a1), make_frag(b0, b1), acc);
    }

#pragma unroll
    for (int r = 0; r < 8; ++r) {
      size_t o = xoff + (size_t)r * (T * H);
      float xp = __builtin_nontemporal_load(xproj + o);            // read-once stream
      float v = tanhf(acc[r] + xp);
      hnext[hbase + (size_t)r * H] = f2bf(v);
      if (OUT_BF16) {
        ((unsigned short*)out_seq)[o] = f2bf(v);                   // re-read by GEMM1
      } else {
        __builtin_nontemporal_store(v, (float*)out_seq + o);       // write-once
      }
      if (t == T - 1) hn_out[hbase + (size_t)r * H] = v;
    }
    xoff += H;

    grid_sync(&ctrl[0], &ctrl[1], (unsigned)nblocks);
  }
}

// ---------------------------------------------------------------------------
extern "C" void kernel_launch(void* const* d_in, const int* in_sizes, int n_in,
                              void* d_out, int out_size, void* d_ws, size_t ws_size,
                              hipStream_t stream) {
  (void)in_sizes; (void)n_in; (void)out_size; (void)ws_size;

  const float* x    = (const float*)d_in[0];
  const float* h0   = (const float*)d_in[1];
  const float* Wxh0 = (const float*)d_in[2];
  const float* Whh0 = (const float*)d_in[3];
  const float* b0   = (const float*)d_in[4];
  const float* Wxh1 = (const float*)d_in[5];
  const float* Whh1 = (const float*)d_in[6];
  const float* b1   = (const float*)d_in[7];
  float* out = (float*)d_out;

  char* ws = (char*)d_ws;
  unsigned*       ctrl   = (unsigned*)(ws + CTRL_OFF);
  unsigned short* xbf    = (unsigned short*)(ws + XBF_OFF);
  unsigned short* wxh0b  = (unsigned short*)(ws + WXH0_OFF);
  unsigned short* whh0b  = (unsigned short*)(ws + WHH0_OFF);
  unsigned short* wxh1b  = (unsigned short*)(ws + WXH1_OFF);
  unsigned short* whh1b  = (unsigned short*)(ws + WHH1_OFF);
  unsigned short* hb0    = (unsigned short*)(ws + HB0_OFF);
  unsigned short* hb1    = (unsigned short*)(ws + HB1_OFF);
  float*          xproj  = (float*)(ws + XPROJ_OFF);
  unsigned short* out0bf = (unsigned short*)(ws + OUT0_OFF);

  float* hn0 = out + OUT1_ELEMS;                  // h_n[0]
  float* hn1 = out + OUT1_ELEMS + (size_t)B * H;  // h_n[1]

  zero_ctrl_kernel<<<1, 32, 0, stream>>>(ctrl);

  // bf16 conversions
  f32_to_bf16_kernel<<<2048, 256, 0, stream>>>(x, xbf, M0 * I);
  f32_to_bf16_kernel<<<256, 256, 0, stream>>>(Wxh0, wxh0b, (size_t)H * I);
  f32_to_bf16_kernel<<<256, 256, 0, stream>>>(Whh0, whh0b, (size_t)H * H);
  f32_to_bf16_kernel<<<256, 256, 0, stream>>>(Wxh1, wxh1b, (size_t)H * H);
  f32_to_bf16_kernel<<<256, 256, 0, stream>>>(Whh1, whh1b, (size_t)H * H);
  f32_to_bf16_kernel<<<64, 256, 0, stream>>>(h0, hb0, (size_t)B * H);                 // -> hb0[0]
  f32_to_bf16_kernel<<<64, 256, 0, stream>>>(h0 + (size_t)B * H, hb1, (size_t)B * H); // -> hb1[0]

  dim3 ggrid(H / 256, (unsigned)(M0 / 128));   // 4 x 256 blocks

  // Layer 0: xproj0 = x @ Wxh0^T + b0 ; then scan (writes out0 in bf16)
  gemm_bias_kernel<<<ggrid, 256, 0, stream>>>(xbf, wxh0b, b0, xproj, I);
  rnn_scan_kernel<true><<<32, 256, 0, stream>>>(xproj, whh0b, hb0, (void*)out0bf,
                                                hn0, ctrl, 32);

  // Layer 1: xproj1 = out0 @ Wxh1^T + b1 (reuse xproj buffer); scan -> d_out f32
  gemm_bias_kernel<<<ggrid, 256, 0, stream>>>(out0bf, wxh1b, b1, xproj, H);
  rnn_scan_kernel<false><<<32, 256, 0, stream>>>(xproj, whh1b, hb1, (void*)out,
                                                 hn1, ctrl, 32);
}
